// ConditionalPerceiverEncoder_68264210202903
// MI455X (gfx1250) — compile-verified, matrix operand order checked by reading, not run on hardware
//
#include <hip/hip_runtime.h>
#include <hip/hip_bf16.h>
#include <math.h>

typedef _Float16 f16;
typedef __attribute__((ext_vector_type(16))) _Float16 v16h;
typedef __attribute__((ext_vector_type(8)))  _Float16 v8h;
typedef __attribute__((ext_vector_type(8)))  float    v8f;
typedef __attribute__((ext_vector_type(4)))  int      v4i;

#define DIM    768
#define NHEAD  8
#define HDIM   96
#define NEXP   8
#define NLAT   256
#define BATCH  32
#define NPATCH 196
#define NVIS   49
#define NMSK   147
#define PKDIM  2304   /* 9*16*16 */
#define GG     14
#define IMGSZ  224

#define BM 128
#define BN 64
#define BK 32
#define LDP 40        /* padded LDS row stride (f16): 80B -> conflict-free b128 reads */

#if __has_builtin(__builtin_amdgcn_global_load_async_to_lds_b128)
#define HAVE_ASYNC_LDS 1
#else
#define HAVE_ASYNC_LDS 0
#endif

__device__ __forceinline__ void async_copy_b128(const f16* gsrc, f16* ldst) {
#if HAVE_ASYNC_LDS
  f16* gnc = const_cast<f16*>(gsrc);
  __builtin_amdgcn_global_load_async_to_lds_b128(
      (__attribute__((address_space(1))) v4i*)gnc,
      (__attribute__((address_space(3))) v4i*)ldst, 0, 0);
#else
  (void)gsrc; (void)ldst;
#endif
}

__device__ __forceinline__ void wait_async0() {
#if HAVE_ASYNC_LDS
#if __has_builtin(__builtin_amdgcn_s_wait_asynccnt)
  __builtin_amdgcn_s_wait_asynccnt(0);
#else
  asm volatile("s_wait_asynccnt 0x0" ::: "memory");
#endif
#endif
}

// ---------------------------------------------------------------- GEMM ------
// C(MxN) = A(MxK, f16 row-major) * B^T (+ epilogue).   REQUIRES K % 32 == 0.
// Edge handling: staged row indices are clamped (always-in-bounds vector
// loads); epilogue stores are guarded. A must be zero-padded in any K-pad
// region (softmax writes the pad zeros for the AV GEMM).
// bMode: 0 = B is f32 NxK row-major (weights, converted to f16 while staging)
//        1 = B is f16 NxK row-major
// cMode: 0 f32 store (+bias) | 1 f16 store (+bias) | 2 f16 gelu(acc+bias)
//        3 f16 head-split QKV | 4 f32 moe accumulate | 5 f16 patch-embed
//        6 f16 head-merge (attn out) | 7 f32 moe init | 8 f16 head-split V^T
struct GemmArgs {
  const f16* A; long zA; int lda;
  const void* B; long zB; int ldb; int bMode;
  const float* bias;
  void* C; long zC; int ldc; int cMode;
  int M, N, K;
  int S, eidx;
  const float* f0; const float* f1; const float* f2;
  const int* i0; const int* i1;
};

__device__ __forceinline__ float gelu_exact(float x) {
  return 0.5f * x * (1.0f + erff(x * 0.70710678118654752f));
}

__global__ __launch_bounds__(256) void gemm_wmma_kernel(GemmArgs g) {
  __shared__ f16 sA[2][BM][LDP];
  __shared__ f16 sB[2][BN][LDP];

  const int tid  = threadIdx.x;
  const int lane = tid & 31;
  const int w    = tid >> 5;
  const int wm   = (w & 3) * 32;   // 4 waves along M
  const int wn   = (w >> 2) * 32;  // 2 waves along N
  const int lo   = (lane & 16) ? 8 : 0;
  const int lr   = lane & 15;
  const int hi8  = (lane & 16) ? 8 : 0;

  const int z  = blockIdx.z;
  const int m0 = blockIdx.y * BM;
  const int n0 = blockIdx.x * BN;

  const f16* Ab = g.A + (long)z * g.zA;

  // ---- staging coordinates (unconditional, index-clamped) ----
  const int arow = tid >> 1, akc = (tid & 1) * 16;   // A: 128 rows x 32k, 16 f16/thread
  const int brow = tid >> 2, bkc = (tid & 3) * 8;    // B: 64 rows x 32k, 8 f16/thread
  int agm = m0 + arow; if (agm > g.M - 1) agm = g.M - 1;
  int bgn = n0 + brow; if (bgn > g.N - 1) bgn = g.N - 1;
  const f16*   aSrc  = Ab + (long)agm * g.lda;
  const f16*   bSrcH = (const f16*)g.B + (long)z * g.zB + (long)bgn * g.ldb;
  const float* bSrcF = (const float*)g.B + (long)z * g.zB + (long)bgn * g.ldb;

  v8f acc[2][2] = {};
  const int nk = g.K / BK;

  // register staging (fallback path, and f32-weight B path)
  v8h pa0, pa1, pb0;
  auto loadA_reg = [&](int k0) {
    pa0 = *(const v8h*)(aSrc + k0 + akc);
    pa1 = *(const v8h*)(aSrc + k0 + akc + 8);
  };
  auto loadB_reg = [&](int k0) {
    if (g.bMode == 1) {
      pb0 = *(const v8h*)(bSrcH + k0 + bkc);
    } else {
      float4 f0 = *(const float4*)(bSrcF + k0 + bkc);
      float4 f1 = *(const float4*)(bSrcF + k0 + bkc + 4);
      pb0[0] = (f16)f0.x; pb0[1] = (f16)f0.y; pb0[2] = (f16)f0.z; pb0[3] = (f16)f0.w;
      pb0[4] = (f16)f1.x; pb0[5] = (f16)f1.y; pb0[6] = (f16)f1.z; pb0[7] = (f16)f1.w;
    }
  };
  auto storeA_reg = [&](int p) {
    *(v8h*)&sA[p][arow][akc]     = pa0;
    *(v8h*)&sA[p][arow][akc + 8] = pa1;
  };
  auto storeB_reg = [&](int p) {
    *(v8h*)&sB[p][brow][bkc] = pb0;
  };

  auto compute = [&](int p) {
    // ---- fragments (ISA 7.12.2: M/N = lane%16, K chunks {lo, 16+lo}) ----
    v16h af[2], bg2[2];
    #pragma unroll
    for (int am = 0; am < 2; ++am) {
      v8h c0 = *(const v8h*)&sA[p][wm + am * 16 + lr][lo];
      v8h c1 = *(const v8h*)&sA[p][wm + am * 16 + lr][16 + lo];
      #pragma unroll
      for (int i = 0; i < 8; ++i) { af[am][i] = c0[i]; af[am][i + 8] = c1[i]; }
    }
    #pragma unroll
    for (int bn = 0; bn < 2; ++bn) {
      v8h c0 = *(const v8h*)&sB[p][wn + bn * 16 + lr][lo];
      v8h c1 = *(const v8h*)&sB[p][wn + bn * 16 + lr][16 + lo];
      #pragma unroll
      for (int i = 0; i < 8; ++i) { bg2[bn][i] = c0[i]; bg2[bn][i + 8] = c1[i]; }
    }
    acc[0][0] = __builtin_amdgcn_wmma_f32_16x16x32_f16(false, af[0], false, bg2[0], (short)0, acc[0][0], false, false);
    acc[0][1] = __builtin_amdgcn_wmma_f32_16x16x32_f16(false, af[0], false, bg2[1], (short)0, acc[0][1], false, false);
    acc[1][0] = __builtin_amdgcn_wmma_f32_16x16x32_f16(false, af[1], false, bg2[0], (short)0, acc[1][0], false, false);
    acc[1][1] = __builtin_amdgcn_wmma_f32_16x16x32_f16(false, af[1], false, bg2[1], (short)0, acc[1][1], false, false);
  };

#if HAVE_ASYNC_LDS
  // ---- async global->LDS staging (ASYNCcnt path); f32 weights keep regs ----
  const bool bAsync = (g.bMode == 1);
  auto stageA_async = [&](int p, int k0) {
    async_copy_b128(aSrc + k0 + akc,     &sA[p][arow][akc]);
    async_copy_b128(aSrc + k0 + akc + 8, &sA[p][arow][akc + 8]);
  };
  auto stageB_async = [&](int p, int k0) {
    async_copy_b128(bSrcH + k0 + bkc, &sB[p][brow][bkc]);
  };

  stageA_async(0, 0);
  if (bAsync) stageB_async(0, 0);
  else { loadB_reg(0); storeB_reg(0); }
  wait_async0();
  __syncthreads();

  int p = 0;
  for (int kt = 0; kt < nk; ++kt) {
    const bool more = (kt + 1) < nk;
    if (more) {
      stageA_async(p ^ 1, (kt + 1) * BK);   // DMA overlaps the WMMAs below
      if (bAsync) stageB_async(p ^ 1, (kt + 1) * BK);
      else loadB_reg((kt + 1) * BK);
    }
    compute(p);
    if (more && !bAsync) storeB_reg(p ^ 1);
    if (more) wait_async0();
    __syncthreads();
    p ^= 1;
  }
#else
  // ---- register staging fallback ----
  loadA_reg(0); loadB_reg(0);
  storeA_reg(0); storeB_reg(0);
  __syncthreads();

  int p = 0;
  for (int kt = 0; kt < nk; ++kt) {
    const bool more = (kt + 1) < nk;
    if (more) { loadA_reg((kt + 1) * BK); loadB_reg((kt + 1) * BK); }
    compute(p);
    if (more) { storeA_reg(p ^ 1); storeB_reg(p ^ 1); }
    __syncthreads();
    p ^= 1;
  }
#endif

  // ---- epilogue (C/D layout: M = r + 8*(lane>=16), N = lane%16) ----
  #pragma unroll
  for (int am = 0; am < 2; ++am) {
    #pragma unroll
    for (int bn = 0; bn < 2; ++bn) {
      int nn = n0 + wn + bn * 16 + lr;
      #pragma unroll
      for (int r = 0; r < 8; ++r) {
        int mm = m0 + wm + am * 16 + hi8 + r;
        if (mm >= g.M || nn >= g.N) continue;
        float v = acc[am][bn][r];
        switch (g.cMode) {
          case 0: {
            float bi = g.bias ? g.bias[nn] : 0.f;
            ((float*)g.C)[(long)z * g.zC + (long)mm * g.ldc + nn] = v + bi;
          } break;
          case 1: {
            float bi = g.bias ? g.bias[nn] : 0.f;
            ((f16*)g.C)[(long)z * g.zC + (long)mm * g.ldc + nn] = (f16)(v + bi);
          } break;
          case 2: {
            ((f16*)g.C)[(long)mm * g.ldc + nn] = (f16)gelu_exact(v + g.bias[nn]);
          } break;
          case 3: {  // QKV head split: out[((b*H+h)*S+s)*HD+d]
            int h = nn / HDIM, d = nn % HDIM;
            int b = mm / g.S, s = mm % g.S;
            ((f16*)g.C)[(((long)b * NHEAD + h) * g.S + s) * HDIM + d] = (f16)(v + g.bias[nn]);
          } break;
          case 4: {  // MoE accumulate
            float cb = g.f0[(long)mm * NEXP + g.eidx];
            ((float*)g.C)[(long)mm * g.ldc + nn] += (v + g.bias[nn]) * cb;
          } break;
          case 5: {  // patch embed: + conv_b + pos_emb[p] + type_emb[type_ids[b]]
            int pp = g.i0[mm];
            int tb = g.i1[mm / NVIS];
            float vv = v + g.f0[nn] + g.f1[(long)pp * DIM + nn] + g.f2[(long)tb * DIM + nn];
            ((f16*)g.C)[(long)mm * g.ldc + nn] = (f16)vv;
          } break;
          case 6: {  // head merge: out[(b*S+m)*D + h*HD + n]
            int b = z / NHEAD, h = z % NHEAD;
            ((f16*)g.C)[((long)(b * g.S + mm)) * DIM + h * HDIM + nn] = (f16)v;
          } break;
          case 7: {  // MoE init
            float cb = g.f0[(long)mm * NEXP + g.eidx];
            ((float*)g.C)[(long)mm * g.ldc + nn] = (v + g.bias[nn]) * cb;
          } break;
          case 8: {  // V^T head split: out[((b*H+h)*HD + d)*ldc + s], ldc = padded S
            int h = nn / HDIM, d = nn % HDIM;
            int b = mm / g.S, s = mm % g.S;
            ((f16*)g.C)[(((long)b * NHEAD + h) * HDIM + d) * g.ldc + s] = (f16)(v + g.bias[nn]);
          } break;
        }
      }
    }
  }
}

// ------------------------------------------------------------ support -------
__global__ void argsort_kernel(const float* noise, int* vis_ws, int* out_idx) {
  __shared__ float sn[NPATCH];
  __shared__ int   so[NPATCH];
  int b = blockIdx.x, tid = threadIdx.x;
  if (tid < NPATCH) sn[tid] = noise[b * NPATCH + tid];
  __syncthreads();
  if (tid < NPATCH) {
    float x = sn[tid];
    int rank = 0;
    for (int j = 0; j < NPATCH; ++j) {
      float y = sn[j];
      rank += (y < x) || (y == x && j < tid);
    }
    so[rank] = tid;
  }
  __syncthreads();
  if (tid < NVIS) {
    int p = so[tid];
    vis_ws[b * NVIS + tid]  = p;
    out_idx[b * NVIS + tid] = p;
  }
  if (tid < NMSK) {
    out_idx[BATCH * NVIS + b * NMSK + tid] = so[NVIS + tid];
  }
}

__global__ void pack_vis_kernel(const float* images, const int* vis, f16* out) {
  long idx = (long)blockIdx.x * blockDim.x + threadIdx.x;
  if (idx >= (long)BATCH * NVIS * PKDIM) return;
  int k = (int)(idx % PKDIM);
  int r = (int)(idx / PKDIM);
  int b = r / NVIS;
  int p = vis[r];
  int gy = p / GG, gx = p % GG;
  int c = k >> 8, rem = k & 255, py = rem >> 4, px = rem & 15;
  float v = images[(((long)b * 3 + c) * IMGSZ + gy * 16 + py) * IMGSZ + gx * 16 + px];
  out[idx] = (f16)v;
}

__global__ void bcast_lat_kernel(const float* latents, float* latf, f16* lat16) {
  long idx = (long)blockIdx.x * blockDim.x + threadIdx.x;
  if (idx >= (long)BATCH * NLAT * DIM) return;
  float v = latents[idx % ((long)NLAT * DIM)];
  latf[idx]  = v;
  lat16[idx] = (f16)v;
}

__global__ __launch_bounds__(256) void softmax_kernel(const float* S, f16* P,
                                                      int cols, int lds, int ldp, float scale) {
  __shared__ float red[256];
  long r = blockIdx.x;
  int tid = threadIdx.x;
  float v = (tid < cols) ? S[r * (long)lds + tid] * scale : -3.4e38f;
  red[tid] = v; __syncthreads();
  for (int o = 128; o; o >>= 1) { if (tid < o) red[tid] = fmaxf(red[tid], red[tid + o]); __syncthreads(); }
  float mx = red[0]; __syncthreads();
  float e = (tid < cols) ? __expf(v - mx) : 0.f;
  red[tid] = e; __syncthreads();
  for (int o = 128; o; o >>= 1) { if (tid < o) red[tid] += red[tid + o]; __syncthreads(); }
  float s = red[0];
  if (tid < ldp) P[r * (long)ldp + tid] = (f16)((tid < cols) ? e / s : 0.f);  // pad cols -> 0
}

__global__ __launch_bounds__(256) void ln_kernel(const float* master, const float* delta,
                                                 const float* gg, const float* bb,
                                                 float* outM, f16* out16) {
  __shared__ float red[256];
  long base = (long)blockIdx.x * DIM;
  int tid = threadIdx.x;
  float x[3]; float s = 0.f;
  #pragma unroll
  for (int i = 0; i < 3; ++i) {
    int c = tid + i * 256;
    x[i] = master[base + c] + delta[base + c];
    s += x[i];
  }
  red[tid] = s; __syncthreads();
  for (int o = 128; o; o >>= 1) { if (tid < o) red[tid] += red[tid + o]; __syncthreads(); }
  float mean = red[0] * (1.f / DIM); __syncthreads();
  float vv = 0.f;
  #pragma unroll
  for (int i = 0; i < 3; ++i) { float d = x[i] - mean; vv += d * d; }
  red[tid] = vv; __syncthreads();
  for (int o = 128; o; o >>= 1) { if (tid < o) red[tid] += red[tid + o]; __syncthreads(); }
  float inv = rsqrtf(red[0] * (1.f / DIM) + 1e-5f);
  #pragma unroll
  for (int i = 0; i < 3; ++i) {
    int c = tid + i * 256;
    float y = (x[i] - mean) * inv * gg[c] + bb[c];
    outM[base + c]  = y;
    out16[base + c] = (f16)y;
  }
}

__global__ __launch_bounds__(256) void router_kernel(const float* x, const float* rw,
                                                     const float* rb, float* comb) {
  int wv = threadIdx.x >> 5, lane = threadIdx.x & 31;
  long t = (long)blockIdx.x * 8 + wv;
  float p[NEXP];
  #pragma unroll
  for (int e = 0; e < NEXP; ++e) p[e] = 0.f;
  for (int c = lane; c < DIM; c += 32) {
    float xv = x[t * DIM + c];
    #pragma unroll
    for (int e = 0; e < NEXP; ++e) p[e] += xv * rw[e * DIM + c];
  }
  #pragma unroll
  for (int e = 0; e < NEXP; ++e) {
    #pragma unroll
    for (int o = 16; o; o >>= 1) p[e] += __shfl_xor(p[e], o);
  }
  if (lane == 0) {
    float mx = -3.4e38f;
    #pragma unroll
    for (int e = 0; e < NEXP; ++e) { p[e] += rb[e]; mx = fmaxf(mx, p[e]); }
    float s = 0.f, pr[NEXP];
    #pragma unroll
    for (int e = 0; e < NEXP; ++e) { pr[e] = __expf(p[e] - mx); s += pr[e]; }
    int i0 = 0;
    #pragma unroll
    for (int e = 1; e < NEXP; ++e) if (pr[e] > pr[i0]) i0 = e;
    int i1 = (i0 == 0) ? 1 : 0;
    #pragma unroll
    for (int e = 0; e < NEXP; ++e) if (e != i0 && pr[e] > pr[i1]) i1 = e;
    float p0 = pr[i0] / s, p1 = pr[i1] / s;
    float e1 = __expf(p1 - p0);          // e0 = 1 (p0 is max)
    float den = 1.f + e1;
    #pragma unroll
    for (int e = 0; e < NEXP; ++e) {
      float cv = (e == i0) ? (1.f / den) : ((e == i1) ? (e1 / den) : 0.f);
      comb[t * NEXP + e] = cv;
    }
  }
}

// ------------------------------------------------------------- host ---------
static inline void launch_gemm(hipStream_t st, const f16* A, long zA, int lda,
                               const void* B, long zB, int ldb, int bMode,
                               const float* bias, void* C, long zC, int ldc, int cMode,
                               int M, int N, int K, int Z, int S, int eidx,
                               const float* f0 = nullptr, const float* f1 = nullptr,
                               const float* f2 = nullptr,
                               const int* i0 = nullptr, const int* i1 = nullptr) {
  GemmArgs g{A, zA, lda, B, zB, ldb, bMode, bias, C, zC, ldc, cMode,
             M, N, K, S, eidx, f0, f1, f2, i0, i1};
  dim3 grid((N + BN - 1) / BN, (M + BM - 1) / BM, Z);
  gemm_wmma_kernel<<<grid, dim3(256), 0, st>>>(g);
}

extern "C" void kernel_launch(void* const* d_in, const int* in_sizes, int n_in,
                              void* d_out, int out_size, void* d_ws, size_t ws_size,
                              hipStream_t stream) {
  const float* images   = (const float*)d_in[0];
  const int*   type_ids = (const int*)d_in[1];
  const float* noise    = (const float*)d_in[2];
  const float* conv_w   = (const float*)d_in[3];
  const float* conv_b   = (const float*)d_in[4];
  const float* pos_emb  = (const float*)d_in[5];
  const float* type_emb = (const float*)d_in[6];
  const float* latents  = (const float*)d_in[7];
  const float* ca_in_w  = (const float*)d_in[8];
  const float* ca_in_b  = (const float*)d_in[9];
  const float* ca_out_w = (const float*)d_in[10];
  const float* ca_out_b = (const float*)d_in[11];
  const float* ca_ln_g  = (const float*)d_in[12];
  const float* ca_ln_b  = (const float*)d_in[13];
  const float* sa_in_w  = (const float*)d_in[14];
  const float* sa_in_b  = (const float*)d_in[15];
  const float* sa_out_w = (const float*)d_in[16];
  const float* sa_out_b = (const float*)d_in[17];
  const float* ln1_g    = (const float*)d_in[18];
  const float* ln1_b    = (const float*)d_in[19];
  const float* router_w = (const float*)d_in[20];
  const float* router_b = (const float*)d_in[21];
  const float* e_w1     = (const float*)d_in[22];
  const float* e_b1     = (const float*)d_in[23];
  const float* e_w2     = (const float*)d_in[24];
  const float* e_b2     = (const float*)d_in[25];
  const float* ln2_g    = (const float*)d_in[26];
  const float* ln2_b    = (const float*)d_in[27];

  char* ws = (char*)d_ws;
  size_t off = 0;
  auto alloc = [&](size_t bytes) -> char* {
    char* p = ws + off;
    off = (off + bytes + 255) & ~(size_t)255;
    return p;
  };
  int*   vis_idx = (int*)  alloc((size_t)BATCH * NVIS * 4);
  f16*   visx16  = (f16*)  alloc((size_t)BATCH * NVIS * PKDIM * 2);
  f16*   vis16   = (f16*)  alloc((size_t)BATCH * NVIS * DIM * 2);
  float* latf    = (float*)alloc((size_t)BATCH * NLAT * DIM * 4);
  f16*   lat16   = (f16*)  alloc((size_t)BATCH * NLAT * DIM * 2);
  f16*   q16     = (f16*)  alloc((size_t)BATCH * NLAT * DIM * 2);
  f16*   k16     = (f16*)  alloc((size_t)BATCH * NLAT * DIM * 2);
  f16*   vt16    = (f16*)  alloc((size_t)BATCH * NHEAD * HDIM * NLAT * 2); // V^T, padded S
  f16*   o16     = (f16*)  alloc((size_t)BATCH * NLAT * DIM * 2);
  float* sc      = (float*)alloc((size_t)BATCH * NHEAD * NLAT * NLAT * 4);
  f16*   pr16    = (f16*)  alloc((size_t)BATCH * NHEAD * NLAT * NLAT * 2);
  float* delta   = (float*)alloc((size_t)BATCH * NLAT * DIM * 4);
  f16*   h16     = (f16*)  alloc((size_t)BATCH * NLAT * 2 * DIM * 2);
  float* comb    = (float*)alloc((size_t)BATCH * NLAT * NEXP * 4);

  float* out_lat = (float*)d_out;
  int*   out_idx = (int*)d_out + (size_t)BATCH * NLAT * DIM;

  const int T  = BATCH * NLAT;   // 8192
  const int TV = BATCH * NVIS;   // 1568
  const float iscale = 0.1020620726159658f;  // 1/sqrt(96)

  // 1. mask argsort -> indices (also straight into d_out tail)
  argsort_kernel<<<dim3(BATCH), dim3(256), 0, stream>>>(noise, vis_idx, out_idx);
  // 2. gather visible patch pixels -> f16
  {
    long n = (long)BATCH * NVIS * PKDIM;
    pack_vis_kernel<<<dim3((unsigned)((n + 255) / 256)), dim3(256), 0, stream>>>(images, vis_idx, visx16);
  }
  // 3. patch embed GEMM (visible rows only) + conv_b + pos + type
  launch_gemm(stream, visx16, 0, PKDIM, conv_w, 0, PKDIM, 0, nullptr,
              vis16, 0, DIM, 5, TV, DIM, PKDIM, 1, 0, 0,
              conv_b, pos_emb, type_emb, vis_idx, type_ids);
  // 4. broadcast latents
  {
    long n = (long)T * DIM;
    bcast_lat_kernel<<<dim3((unsigned)((n + 255) / 256)), dim3(256), 0, stream>>>(latents, latf, lat16);
  }
  // 5. cross attention (kv = 49 visible tokens, scores ld padded to 64)
  launch_gemm(stream, lat16, 0, DIM, ca_in_w, 0, DIM, 0, ca_in_b,
              q16, 0, 0, 3, T, DIM, DIM, 1, NLAT, 0);
  launch_gemm(stream, vis16, 0, DIM, ca_in_w + (size_t)DIM * DIM, 0, DIM, 0, ca_in_b + DIM,
              k16, 0, 0, 3, TV, DIM, DIM, 1, NVIS, 0);
  launch_gemm(stream, vis16, 0, DIM, ca_in_w + (size_t)2 * DIM * DIM, 0, DIM, 0, ca_in_b + 2 * DIM,
              vt16, 0, 64, 8, TV, DIM, DIM, 1, NVIS, 0);
  launch_gemm(stream, q16, (long)NLAT * HDIM, HDIM, k16, (long)NVIS * HDIM, HDIM, 1, nullptr,
              sc, (long)NLAT * 64, 64, 0, NLAT, NVIS, HDIM, BATCH * NHEAD, 0, 0);
  softmax_kernel<<<dim3(BATCH * NHEAD * NLAT), dim3(256), 0, stream>>>(sc, pr16, NVIS, 64, 64, iscale);
  launch_gemm(stream, pr16, (long)NLAT * 64, 64, vt16, (long)HDIM * 64, 64, 1, nullptr,
              o16, 0, 0, 6, NLAT, HDIM, 64, BATCH * NHEAD, NLAT, 0);
  launch_gemm(stream, o16, 0, DIM, ca_out_w, 0, DIM, 0, ca_out_b,
              delta, 0, DIM, 0, T, DIM, DIM, 1, 0, 0);
  ln_kernel<<<dim3(T), dim3(256), 0, stream>>>(latf, delta, ca_ln_g, ca_ln_b, latf, lat16);

  // 6. transformer layers
  for (int l = 0; l < 6; ++l) {
    const float* inw = sa_in_w + (size_t)l * 3 * DIM * DIM;
    const float* inb = sa_in_b + (size_t)l * 3 * DIM;
    launch_gemm(stream, lat16, 0, DIM, inw, 0, DIM, 0, inb,
                q16, 0, 0, 3, T, DIM, DIM, 1, NLAT, 0);
    launch_gemm(stream, lat16, 0, DIM, inw + (size_t)DIM * DIM, 0, DIM, 0, inb + DIM,
                k16, 0, 0, 3, T, DIM, DIM, 1, NLAT, 0);
    launch_gemm(stream, lat16, 0, DIM, inw + (size_t)2 * DIM * DIM, 0, DIM, 0, inb + 2 * DIM,
                vt16, 0, NLAT, 8, T, DIM, DIM, 1, NLAT, 0);
    launch_gemm(stream, q16, (long)NLAT * HDIM, HDIM, k16, (long)NLAT * HDIM, HDIM, 1, nullptr,
                sc, (long)NLAT * NLAT, NLAT, 0, NLAT, NLAT, HDIM, BATCH * NHEAD, 0, 0);
    softmax_kernel<<<dim3(BATCH * NHEAD * NLAT), dim3(256), 0, stream>>>(sc, pr16, NLAT, NLAT, NLAT, iscale);
    launch_gemm(stream, pr16, (long)NLAT * NLAT, NLAT, vt16, (long)HDIM * NLAT, NLAT, 1, nullptr,
                o16, 0, 0, 6, NLAT, HDIM, NLAT, BATCH * NHEAD, NLAT, 0);
    launch_gemm(stream, o16, 0, DIM, sa_out_w + (size_t)l * DIM * DIM, 0, DIM, 0,
                sa_out_b + (size_t)l * DIM, delta, 0, DIM, 0, T, DIM, DIM, 1, 0, 0);
    ln_kernel<<<dim3(T), dim3(256), 0, stream>>>(latf, delta,
                                                 ln1_g + (size_t)l * DIM, ln1_b + (size_t)l * DIM,
                                                 latf, lat16);
    // MoE
    router_kernel<<<dim3(T / 8), dim3(256), 0, stream>>>(latf,
                                                         router_w + (size_t)l * NEXP * DIM,
                                                         router_b + (size_t)l * NEXP, comb);
    for (int e = 0; e < NEXP; ++e) {
      const float* w1 = e_w1 + ((size_t)l * NEXP + e) * 2 * DIM * DIM;
      const float* b1 = e_b1 + ((size_t)l * NEXP + e) * 2 * DIM;
      const float* w2 = e_w2 + ((size_t)l * NEXP + e) * DIM * 2 * DIM;
      const float* b2 = e_b2 + ((size_t)l * NEXP + e) * DIM;
      launch_gemm(stream, lat16, 0, DIM, w1, 0, DIM, 0, b1,
                  h16, 0, 2 * DIM, 2, T, 2 * DIM, DIM, 1, 0, 0);
      launch_gemm(stream, h16, 0, 2 * DIM, w2, 0, 2 * DIM, 0, b2,
                  delta, 0, DIM, (e == 0) ? 7 : 4, T, DIM, 2 * DIM, 1, 0, e, comb);
    }
    float* outM = (l == 5) ? out_lat : latf;
    ln_kernel<<<dim3(T), dim3(256), 0, stream>>>(latf, delta,
                                                 ln2_g + (size_t)l * DIM, ln2_b + (size_t)l * DIM,
                                                 outM, lat16);
  }
}
